// FAME_47038481826249
// MI455X (gfx1250) — compile-verified, matrix operand order checked by reading, not run on hardware
//
#include <hip/hip_runtime.h>
#include <cmath>

#define B_     64
#define C_     3
#define T_     16
#define H_     112
#define W_     112
#define HW_    (H_*W_)
#define CTHW_  (C_*T_*HW_)
#define NBINS_ 1000
#define KFG_   6272    /* int(0.5*HW)  */
#define KBG_   1254    /* int(0.1*HW)  */
#define NSEL_  6272    /* int(BETA*HW) */
#define EPS_   1e-8f
#define KPAD_  128
#define TWO_PI_ 6.283185307179586f

typedef __attribute__((ext_vector_type(16))) _Float16 v16h;
typedef __attribute__((ext_vector_type(8)))  float    v8f;

union Frag16 { v16h v; _Float16 e[16]; };
union Acc8   { v8f   v; float    e[8]; };

// ---------------------------------------------------------------------------
// Build banded blur matrix A (112x128, f16, reflect-padded 11-tap gaussian).
// A[n][j] = sum of g[t] over taps t where reflect(n+t-5) == j ; j>=112 -> 0.
// ---------------------------------------------------------------------------
__global__ void build_A(_Float16* __restrict__ Ah) {
  const float sigma = 11.0f / 3.0f;
  float g[11]; float gs = 0.f;
  #pragma unroll
  for (int t = 0; t < 11; ++t) {
    float x = (float)(t - 5);
    g[t] = __expf(-(x * x) / (2.f * sigma * sigma));
    gs += g[t];
  }
  const float inv = 1.f / gs;
  for (int idx = threadIdx.x; idx < H_ * KPAD_; idx += blockDim.x) {
    int n = idx / KPAD_, j = idx % KPAD_;
    float acc = 0.f;
    if (j < W_) {
      #pragma unroll
      for (int t = 0; t < 11; ++t) {
        int ii = n + t - 5;
        if (ii < 0) ii = -ii;
        if (ii > 111) ii = 222 - ii;
        if (ii == j) acc += g[t] * inv;
      }
    }
    Ah[idx] = (_Float16)acc;
  }
}

// ---------------------------------------------------------------------------
// Fused single pass over the 154MB video: temporal abs-diff + temporal mean.
// ---------------------------------------------------------------------------
__global__ void diff_mean(const float* __restrict__ vin,
                          float* __restrict__ imdiff,
                          float* __restrict__ img) {
  int p = blockIdx.x * blockDim.x + threadIdx.x;
  if (p >= B_ * HW_) return;
  int b = p / HW_, hw = p % HW_;
  const float* base = vin + (size_t)b * CTHW_ + hw;
  float diff = 0.f;
  #pragma unroll
  for (int c = 0; c < C_; ++c) {
    const float* cb = base + (size_t)c * T_ * HW_;
    float prev = cb[0];
    float s = prev;
    for (int t = 1; t < T_; ++t) {
      float cur = cb[(size_t)t * HW_];
      s += cur;
      diff += fabsf(cur - prev);
      prev = cur;
    }
    img[((size_t)b * C_ + c) * HW_ + hw] = s * (1.f / T_);
  }
  imdiff[p] = diff * (1.f / (T_ - 1));
}

// ---------------------------------------------------------------------------
// out = (in . A^T)^T via WMMA f16->f32.  Applying twice gives A . in . A^T,
// i.e. the full separable reflect-padded gaussian blur.
// One workgroup per image; X and A staged to LDS as f16 (K padded to 128).
// Fragment layouts per CDNA5 ISA 7.12.2 (wave32).
// ---------------------------------------------------------------------------
__global__ void __launch_bounds__(256) blur_T(const float* __restrict__ in,
                                              float* __restrict__ out,
                                              const _Float16* __restrict__ Ah) {
  __shared__ _Float16 sX[H_ * KPAD_];
  __shared__ _Float16 sA[H_ * KPAD_];
  const int b = blockIdx.x;
  const float* src = in + (size_t)b * HW_;
  for (int idx = threadIdx.x; idx < H_ * KPAD_; idx += 256) {
    int r = idx >> 7, c = idx & 127;
    sX[idx] = (c < W_) ? (_Float16)src[r * W_ + c] : (_Float16)0.f;
    sA[idx] = Ah[idx];
  }
  __syncthreads();

  const int wave = threadIdx.x >> 5;
  const int lane = threadIdx.x & 31;
  const int hi   = lane >> 4;    // which half-wave (K +8 group / M +8 group)
  const int l15  = lane & 15;    // M for A-frag, N for B/C/D frags

  for (int tile = wave; tile < 49; tile += 8) {
    const int tm = (tile / 7) * 16;   // row block of `in`   (M)
    const int tn = (tile % 7) * 16;   // row block of A      (N)
    const _Float16* xr = &sX[(tm + l15) * KPAD_];
    const _Float16* ar = &sA[(tn + l15) * KPAD_];
    Acc8 acc;
    #pragma unroll
    for (int v = 0; v < 8; ++v) acc.e[v] = 0.f;

    #pragma unroll
    for (int kb = 0; kb < KPAD_; kb += 32) {
      Frag16 fa, fb;
      #pragma unroll
      for (int v = 0; v < 8; ++v) {
        // ISA 7.12.2 16-bit A layout: VGPR v holds K = {2v, 2v+1} (+16 for
        // v>=4), hi half-wave offsets K by +8. B uses the same K striping
        // with N = lane&15; here B[k][n] = A[tn+n][k] so both are row reads.
        const int kk = kb + ((v < 4) ? (2 * v) : (16 + 2 * (v - 4))) + hi * 8;
        fa.e[2 * v]     = xr[kk];
        fa.e[2 * v + 1] = xr[kk + 1];
        fb.e[2 * v]     = ar[kk];
        fb.e[2 * v + 1] = ar[kk + 1];
      }
      acc.v = __builtin_amdgcn_wmma_f32_16x16x32_f16(
          /*neg_a=*/false, fa.v, /*neg_b=*/false, fb.v,
          /*c_mod=*/(short)0, acc.v, /*reuse_a=*/false, /*reuse_b=*/false);
    }
    // D[M][N]: M = v + 8*hi, N = l15. Store transposed: out[n][m] = D[m][n].
    float* orow = out + (size_t)b * HW_ + (tn + l15) * W_ + tm + hi * 8;
    #pragma unroll
    for (int v = 0; v < 8; ++v) orow[v] = acc.e[v];
  }
}

// ---------------------------------------------------------------------------
// Per-image (x - min) / (max - min + eps), one workgroup per image.
// ---------------------------------------------------------------------------
__global__ void __launch_bounds__(256) minmax_norm(float* __restrict__ buf) {
  __shared__ float rmn[256], rmx[256];
  const int b = blockIdx.x, tid = threadIdx.x;
  float* m = buf + (size_t)b * HW_;
  float mn = 3.4e38f, mx = -3.4e38f;
  for (int p = tid; p < HW_; p += 256) {
    float v = m[p];
    mn = fminf(mn, v);
    mx = fmaxf(mx, v);
  }
  rmn[tid] = mn; rmx[tid] = mx;
  __syncthreads();
  for (int s = 128; s > 0; s >>= 1) {
    if (tid < s) {
      rmn[tid] = fminf(rmn[tid], rmn[tid + s]);
      rmx[tid] = fmaxf(rmx[tid], rmx[tid + s]);
    }
    __syncthreads();
  }
  const float lo = rmn[0];
  const float scale = 1.f / (rmx[0] - rmn[0] + EPS_);
  for (int p = tid; p < HW_; p += 256) m[p] = (m[p] - lo) * scale;
}

// ---------------------------------------------------------------------------
// RGB -> HSV -> 1000-bin color map index per pixel (replicates reference).
// ---------------------------------------------------------------------------
__global__ void hsv_cmap(const float* __restrict__ img, int* __restrict__ cmap) {
  int p = blockIdx.x * blockDim.x + threadIdx.x;
  if (p >= B_ * HW_) return;
  int b = p / HW_, hw = p % HW_;
  const float* ib = img + (size_t)b * C_ * HW_ + hw;
  float r = ib[0], g = ib[HW_], bl = ib[2 * HW_];
  float maxc = fmaxf(r, fmaxf(g, bl));
  float minc = fminf(r, fminf(g, bl));
  float v = maxc;
  float delta = maxc - minc;
  float s = delta / (maxc + EPS_);
  float dc = (delta == 0.f) ? 1.f : delta;
  float rc = (maxc - r) / dc, gc = (maxc - g) / dc, bc = (maxc - bl) / dc;
  float h = (r >= g && r >= bl) ? (bc - gc)
           : ((g >= bl) ? (2.f + rc - bc) : (4.f + gc - rc));
  h = h * (1.f / 6.f);
  h -= floorf(h);          // python % 1.0
  h *= TWO_PI_;
  float hx = (s * __cosf(h * TWO_PI_) + 1.f) * 0.5f;   // as in reference
  float hy = (s * __sinf(h * TWO_PI_) + 1.f) * 0.5f;
  float hb = rintf(hx * (float)(10 - 1) + 1.f);
  float sb = rintf(hy * (float)(10 - 1) + 1.f);
  float vb = rintf(v  * (float)(10 - 1) + 1.f);
  int cm = (int)(hb + (sb - 1.f) * 10.f + (vb - 1.f) * 100.f);
  cm = cm < 0 ? 0 : (cm > NBINS_ - 1 ? NBINS_ - 1 : cm);
  cmap[p] = cm;
}

__device__ __forceinline__ int block_reduce_sum(int val, int* red, int tid) {
  red[tid] = val;
  __syncthreads();
  for (int s = 128; s > 0; s >>= 1) {
    if (tid < s) red[tid] += red[tid + s];
    __syncthreads();
  }
  int r = red[0];
  __syncthreads();
  return r;
}

// ---------------------------------------------------------------------------
// Per-image: top-k thresholds on mask (bisection on normalized [0,1] values),
// LDS fg/bg color histograms, refine = pf / (pf + pb).
// ---------------------------------------------------------------------------
__global__ void __launch_bounds__(256) seg_kernel(const float* __restrict__ mask,
                                                  const int* __restrict__ cmap,
                                                  float* __restrict__ refine) {
  __shared__ int hfg[NBINS_];
  __shared__ int hbg[NBINS_];
  __shared__ int red[256];
  const int b = blockIdx.x, tid = threadIdx.x;
  const float* m = mask + (size_t)b * HW_;
  const int* cm = cmap + (size_t)b * HW_;

  // fg threshold: count(m >= t) ~ KFG_
  float lo = -0.01f, hi = 1.01f;
  for (int it = 0; it < 24; ++it) {
    float mid = 0.5f * (lo + hi);
    int c = 0;
    for (int p = tid; p < HW_; p += 256) c += (m[p] >= mid) ? 1 : 0;
    int cnt = block_reduce_sum(c, red, tid);
    if (cnt > KFG_) lo = mid; else hi = mid;
  }
  const float thf = lo;

  // bg threshold: count(m <= t) ~ KBG_
  lo = -0.01f; hi = 1.01f;
  for (int it = 0; it < 24; ++it) {
    float mid = 0.5f * (lo + hi);
    int c = 0;
    for (int p = tid; p < HW_; p += 256) c += (m[p] <= mid) ? 1 : 0;
    int cnt = block_reduce_sum(c, red, tid);
    if (cnt > KBG_) hi = mid; else lo = mid;
  }
  const float thb = hi;

  for (int i = tid; i < NBINS_; i += 256) { hfg[i] = 0; hbg[i] = 0; }
  __syncthreads();
  for (int p = tid; p < HW_; p += 256) {
    float mv = m[p];
    int c = cm[p];
    if (mv >= thf) atomicAdd(&hfg[c], 1);
    if (mv <= thb) atomicAdd(&hbg[c], 1);
  }
  __syncthreads();

  int sf = 0, sb = 0;
  for (int i = tid; i < NBINS_; i += 256) { sf += hfg[i]; sb += hbg[i]; }
  int nfg = block_reduce_sum(sf, red, tid);
  int nbg = block_reduce_sum(sb, red, tid);
  const float invf = 1.f / ((float)nfg + EPS_);
  const float invb = 1.f / ((float)nbg + (float)NBINS_ + EPS_);
  for (int p = tid; p < HW_; p += 256) {
    int c = cm[p];
    float pf = (float)hfg[c] * invf;
    float pb = ((float)hbg[c] + 1.f) * invb;
    refine[(size_t)b * HW_ + p] = pf / (pf + pb);
  }
}

// ---------------------------------------------------------------------------
// Final top-NSEL_ threshold per image.
// ---------------------------------------------------------------------------
__global__ void __launch_bounds__(256) topk_threshold(const float* __restrict__ mf,
                                                      float* __restrict__ thr) {
  __shared__ int red[256];
  const int b = blockIdx.x, tid = threadIdx.x;
  const float* m = mf + (size_t)b * HW_;
  float lo = -0.01f, hi = 1.01f;
  for (int it = 0; it < 24; ++it) {
    float mid = 0.5f * (lo + hi);
    int c = 0;
    for (int p = tid; p < HW_; p += 256) c += (m[p] >= mid) ? 1 : 0;
    int cnt = block_reduce_sum(c, red, tid);
    if (cnt > NSEL_) lo = mid; else hi = mid;
  }
  if (tid == 0) thr[b] = lo;
}

// ---------------------------------------------------------------------------
// Streaming blend: out = (m >= thr[b]) ? v : v[perm].  2R + 1W of 154MB.
// float4 vectorized; software prefetch (global_prefetch_b8) on the stream.
// ---------------------------------------------------------------------------
__global__ void blend(const float* __restrict__ vin, const int* __restrict__ perm,
                      const float* __restrict__ m, const float* __restrict__ thr,
                      float* __restrict__ out) {
  size_t i = (size_t)blockIdx.x * blockDim.x + threadIdx.x;
  const size_t total4 = (size_t)B_ * CTHW_ / 4;
  if (i >= total4) return;
  size_t e = i * 4;
  int b = (int)(e / CTHW_);
  int rem = (int)(e % CTHW_);
  int hw = rem % HW_;
  __builtin_prefetch(vin + e + 8192, 0, 1);
  const float4 vo = *(const float4*)(vin + e);
  const float4 vp = *(const float4*)(vin + (size_t)perm[b] * CTHW_ + rem);
  const float4 mv = *(const float4*)(m + (size_t)b * HW_ + hw);
  const float th = thr[b];
  float4 o;
  o.x = (mv.x >= th) ? vo.x : vp.x;
  o.y = (mv.y >= th) ? vo.y : vp.y;
  o.z = (mv.z >= th) ? vo.z : vp.z;
  o.w = (mv.w >= th) ? vo.w : vp.w;
  *(float4*)(out + e) = o;
}

// ---------------------------------------------------------------------------
extern "C" void kernel_launch(void* const* d_in, const int* in_sizes, int n_in,
                              void* d_out, int out_size, void* d_ws, size_t ws_size,
                              hipStream_t stream) {
  (void)in_sizes; (void)n_in; (void)out_size; (void)ws_size;
  const float* vin = (const float*)d_in[0];
  const int*   perm = (const int*)d_in[1];
  float* out = (float*)d_out;
  char* ws = (char*)d_ws;

  const size_t SZ = (size_t)B_ * HW_ * sizeof(float);   // 3,211,264 B
  _Float16* Ah     = (_Float16*)(ws);                   // 28,672 B
  float*    thr    = (float*)(ws + 28672);              // 256 B
  float*    imdiff = (float*)(ws + 28928);
  float*    img    = (float*)(ws + 28928 + SZ);         // 3 * SZ
  int*      cmapb  = (int*)  (ws + 28928 + 4 * SZ);
  float*    tmp    = (float*)(ws + 28928 + 5 * SZ);
  float*    maskb  = (float*)(ws + 28928 + 6 * SZ);
  float*    refb   = (float*)(ws + 28928 + 7 * SZ);
  float*    mfin   = (float*)(ws + 28928 + 8 * SZ);

  build_A<<<1, 128, 0, stream>>>(Ah);
  diff_mean<<<(B_ * HW_ + 255) / 256, 256, 0, stream>>>(vin, imdiff, img);

  // mask = norm(blur(im_diff))   [blur = two WMMA GEMM passes]
  blur_T<<<B_, 256, 0, stream>>>(imdiff, tmp, Ah);
  blur_T<<<B_, 256, 0, stream>>>(tmp, maskb, Ah);
  minmax_norm<<<B_, 256, 0, stream>>>(maskb);

  hsv_cmap<<<(B_ * HW_ + 255) / 256, 256, 0, stream>>>(img, cmapb);
  seg_kernel<<<B_, 256, 0, stream>>>(maskb, cmapb, refb);

  // m = norm(blur(refine))
  blur_T<<<B_, 256, 0, stream>>>(refb, tmp, Ah);
  blur_T<<<B_, 256, 0, stream>>>(tmp, mfin, Ah);
  minmax_norm<<<B_, 256, 0, stream>>>(mfin);

  topk_threshold<<<B_, 256, 0, stream>>>(mfin, thr);

  const size_t total4 = (size_t)B_ * CTHW_ / 4;
  blend<<<(unsigned)((total4 + 255) / 256), 256, 0, stream>>>(vin, perm, mfin, thr, out);
}
